// SSM_78477642432794
// MI455X (gfx1250) — matrix operand order repaired and checked by reading, not compiled, not run on hardware
//
#include <hip/hip_runtime.h>
#include <hip/hip_bf16.h>
#include <math.h>

// ---------------------------------------------------------------------------
// Selective SSM (Mamba-style scan), fused for MI455X (gfx1250, wave32).
//   B=4, L=2048, D=768, N=16
//   1) proj_kernel: WMMA (v_wmma_f32_16x16x32_f16) computes Bp, C in 16x16
//      tiles; x chunks are staged into LDS with double-buffered async
//      global->LDS DMA (ASYNCcnt); Wb/Wc live in LDS as f16.
//   2) chunked diagonal scan (3 phases); Bp/C/s1 tiles staged into LDS.
//   deltaA / deltaB_u (100M elems) are never materialized: ~50MB traffic.
// ---------------------------------------------------------------------------

typedef __attribute__((ext_vector_type(16))) _Float16 v16h;
typedef __attribute__((ext_vector_type(8)))  float    v8f;
typedef int v4i __attribute__((vector_size(4 * sizeof(int))));

constexpr int BB = 4;
constexpr int LL = 2048;
constexpr int DD = 768;
constexpr int NN = 16;

#define GLOBAL_AS __attribute__((address_space(1)))
#define LDS_AS    __attribute__((address_space(3)))

#if __has_builtin(__builtin_amdgcn_global_load_async_to_lds_b128)
#define HAVE_ASYNC_LDS 1
#else
#define HAVE_ASYNC_LDS 0
#endif

// 16B global -> LDS copy: async DMA when available, else direct.
__device__ __forceinline__ void g2l_b128(const float* g, float* l) {
#if HAVE_ASYNC_LDS
    __builtin_amdgcn_global_load_async_to_lds_b128(
        (GLOBAL_AS v4i*)g, (LDS_AS v4i*)l, 0, 0);
#else
    const float4 v = *(const float4*)g;
    *(float4*)l = v;
#endif
}

__device__ __forceinline__ void async_wait0() {
#if HAVE_ASYNC_LDS
#if __has_builtin(__builtin_amdgcn_s_wait_asynccnt)
    __builtin_amdgcn_s_wait_asynccnt(0);
#else
    asm volatile("s_wait_asynccnt 0x0" ::: "memory");
#endif
#endif
}

__device__ __forceinline__ float softplusf(float z) {
    return (z > 20.0f) ? z : log1pf(expf(z));
}

// ---------------------------------------------------------------------------
// Projection kernel. Block = 256 threads = 8 waves = 8 row-tiles (128 rows).
// A fragment (16x32 f16): lane L<16: row M=L, K in {0..7,16..23};
//                         lane L>=16: row M=L-16, K in {8..15,24..31}.
// B fragment (32x16 f16): lane L: col N=L&15, K in {0..7,16..23} (L<16) /
//                         {8..15,24..31} (L>=16), 2 halves per VGPR.
// C/D (16x16 f32): VGPR r: lanes 0-15 -> (M=r, N=lane),
//                          lanes 16-31 -> (M=r+8, N=lane-16).
// ---------------------------------------------------------------------------
__global__ __launch_bounds__(256) void proj_kernel(
    const float* __restrict__ x,
    const float* __restrict__ Wb, const float* __restrict__ bb,
    const float* __restrict__ Wc, const float* __restrict__ bc,
    const float* __restrict__ W1, const float* __restrict__ b1,
    float* __restrict__ Bp_out, float* __restrict__ C_out,
    float* __restrict__ s1_out)
{
    constexpr int RPB    = 128;  // rows per block
    constexpr int XPITCH = 36;   // padded floats per staged x row (16B align)
    constexpr int WPITCH = 776;  // padded halfs per weight row (bank spread)

    __shared__ __align__(16) float    sx [2 * RPB * XPITCH];   // 36 KB
    __shared__ __align__(16) _Float16 sWb[16 * WPITCH];        // 24.25 KB
    __shared__ __align__(16) _Float16 sWc[16 * WPITCH];        // 24.25 KB
    __shared__ __align__(16) float    sW1[DD];                 // 3 KB

    const int tid   = threadIdx.x;
    const int wv    = tid >> 5;
    const int lane  = tid & 31;
    const int mrow  = lane & 15;
    const bool hi   = lane >= 16;
    const int kA    = hi ? 8  : 0;
    const int kB    = hi ? 8  : 0;
    const int r0blk = blockIdx.x * RPB;
    const int r0    = r0blk + wv * 16;       // first global row of this tile
    const int lrow  = wv * 16 + mrow;        // this lane's local row

    // ---- preload weights into LDS (f16) + W1 (f32) ------------------------
    for (int idx = tid; idx < 16 * DD; idx += 256) {
        const int n = idx / DD;
        const int k = idx - n * DD;
        sWb[n * WPITCH + k] = (_Float16)Wb[idx];
        sWc[n * WPITCH + k] = (_Float16)Wc[idx];
    }
    for (int k = tid; k < DD; k += 256) sW1[k] = W1[k];

    // ---- stage one 128x32 f32 x-chunk into LDS ----------------------------
    auto stage_x = [&](int k0, int bufi) {
        // 1024 b128 segments, 4 per thread, coalesced rows of 32 floats
        #pragma unroll
        for (int i = 0; i < 4; ++i) {
            const int s   = tid + 256 * i;
            const int row = s >> 3;
            const int col = (s & 7) * 4;
            g2l_b128(x + (size_t)(r0blk + row) * DD + k0 + col,
                     sx + bufi * RPB * XPITCH + row * XPITCH + col);
        }
    };

    stage_x(0, 0);
    async_wait0();
    __syncthreads();

    v8f accB = {}, accC = {};
    float s1p = 0.0f;
    int buf = 0;

    for (int k0 = 0; k0 < DD; k0 += 32) {
        const bool more = (k0 + 32) < DD;
        if (more) stage_x(k0 + 32, buf ^ 1);   // DMA overlaps compute below

        // ---- A fragment from LDS -----------------------------------------
        const float* xr = sx + buf * RPB * XPITCH + lrow * XPITCH;
        float xf[16];
        #pragma unroll
        for (int j = 0; j < 8; ++j) {
            xf[j]     = xr[kA + j];
            xf[8 + j] = xr[kA + 16 + j];
        }
        v16h a;
        #pragma unroll
        for (int j = 0; j < 16; ++j) a[j] = (_Float16)xf[j];

        // ---- s1 partial dot with W1 over this lane's K subset ------------
        #pragma unroll
        for (int j = 0; j < 8; ++j) {
            s1p = fmaf(xf[j],     sW1[k0 + kA + j],      s1p);
            s1p = fmaf(xf[8 + j], sW1[k0 + kA + 16 + j], s1p);
        }

        // ---- B fragments from LDS ----------------------------------------
        const _Float16* wbp = sWb + mrow * WPITCH + k0 + kB;
        const _Float16* wcp = sWc + mrow * WPITCH + k0 + kB;
        v16h fb, fc;
        #pragma unroll
        for (int j = 0; j < 16; ++j) { fb[j] = wbp[j < 8 ? j : j + 8]; fc[j] = wcp[j < 8 ? j : j + 8]; }

        accB = __builtin_amdgcn_wmma_f32_16x16x32_f16(
                   false, a, false, fb, (short)0, accB, false, false);
        accC = __builtin_amdgcn_wmma_f32_16x16x32_f16(
                   false, a, false, fc, (short)0, accC, false, false);

        // publish next chunk + prove all waves finished reading `buf`
        if (more) async_wait0();
        __syncthreads();
        buf ^= 1;
    }

    // s1: lane L and lane L^16 cover disjoint K of the same row
    const float s1o = __shfl_xor(s1p, 16, 32);
    if (!hi) s1_out[r0 + mrow] = s1p + s1o + b1[0];

    #pragma unroll
    for (int r = 0; r < 8; ++r) {
        const int m = hi ? (r + 8) : r;
        const int n = mrow;
        Bp_out[(size_t)(r0 + m) * NN + n] = accB[r] + bb[n];
        C_out [(size_t)(r0 + m) * NN + n] = accC[r] + bc[n];
    }
}

// ---------------------------------------------------------------------------
// Phase 1: per-chunk local scan (h0=0) -> S, decay product -> P.
// grid = B * NC * (D/256); thread owns channel d; Bp/s1 tiles staged in LDS.
// ---------------------------------------------------------------------------
__global__ __launch_bounds__(256) void scan_phase1(
    const float* __restrict__ x, const float* __restrict__ A_log,
    const float* __restrict__ Wd, const float* __restrict__ bd,
    const float* __restrict__ Bp, const float* __restrict__ s1,
    float* __restrict__ Pout, float* __restrict__ Sout, int NC)
{
    constexpr int TILE = 128;
    __shared__ __align__(16) float sBp[TILE * NN];
    __shared__ __align__(16) float sS1[TILE];

    const int CL = LL / NC;
    int gid  = blockIdx.x;
    const int dblk = gid % (DD / 256); gid /= (DD / 256);
    const int c    = gid % NC;
    const int b    = gid / NC;
    const int d    = dblk * 256 + threadIdx.x;

    float an[NN], h[NN], p[NN];
    #pragma unroll
    for (int n = 0; n < NN; ++n) {
        an[n] = -expf(A_log[(size_t)d * NN + n]);
        h[n]  = 0.0f;
        p[n]  = 1.0f;
    }
    const float wd  = Wd[d];
    const float bdv = bd[d];

    const size_t l0g = (size_t)b * LL + (size_t)c * CL;
    const float* xp  = x + l0g * DD + d;

    for (int t0 = 0; t0 < CL; t0 += TILE) {
        const float* bsrc = Bp + (l0g + t0) * NN;
        for (int s = threadIdx.x; s < TILE * NN / 4; s += 256)
            g2l_b128(bsrc + 4 * s, sBp + 4 * s);
        if (threadIdx.x < TILE / 4)
            g2l_b128(s1 + l0g + t0 + 4 * threadIdx.x, sS1 + 4 * threadIdx.x);
        async_wait0();
        __syncthreads();

        for (int l = 0; l < TILE; ++l) {
            const int lg = t0 + l;
            if (lg + 8 < CL) __builtin_prefetch(xp + (size_t)(lg + 8) * DD, 0, 0);
            const float xv    = xp[(size_t)lg * DD];
            const float delta = softplusf(fmaf(sS1[l], wd, bdv));
            const float dx    = delta * xv;
            #pragma unroll
            for (int n = 0; n < NN; ++n) {
                const float dA = expf(delta * an[n]);
                const float u  = dx * sBp[l * NN + n];
                h[n] = fmaf(dA, h[n], u);
                p[n] *= dA;
            }
        }
        __syncthreads();   // before next tile's LDS overwrite
    }

    const size_t base = (((size_t)b * NC + c) * DD + d) * NN;
    #pragma unroll
    for (int n = 0; n < NN; ++n) {
        Sout[base + n] = h[n];
        Pout[base + n] = p[n];
    }
}

// ---------------------------------------------------------------------------
// Phase 2: cross-chunk combine. One thread per (b,d,n) walks NC chunks.
// ---------------------------------------------------------------------------
__global__ __launch_bounds__(256) void scan_phase2(
    const float* __restrict__ P, const float* __restrict__ S,
    float* __restrict__ E, int NC)
{
    const int t = blockIdx.x * blockDim.x + threadIdx.x;
    if (t >= BB * DD * NN) return;
    const int n   = t % NN;
    const int rem = t / NN;
    const int d   = rem % DD;
    const int b   = rem / DD;

    float e = 0.0f;
    for (int c = 0; c < NC; ++c) {
        const size_t idx = (((size_t)b * NC + c) * DD + d) * NN + n;
        E[idx] = e;
        e = fmaf(P[idx], e, S[idx]);
    }
}

// ---------------------------------------------------------------------------
// Phase 3: replay chunks from init state E, emit y (coalesced over d).
// ---------------------------------------------------------------------------
__global__ __launch_bounds__(256) void scan_phase3(
    const float* __restrict__ x, const float* __restrict__ A_log,
    const float* __restrict__ Wd, const float* __restrict__ bd,
    const float* __restrict__ Bp, const float* __restrict__ Cc,
    const float* __restrict__ s1, const float* __restrict__ E,
    float* __restrict__ y, int NC)
{
    constexpr int TILE = 128;
    __shared__ __align__(16) float sBp[TILE * NN];
    __shared__ __align__(16) float sC [TILE * NN];
    __shared__ __align__(16) float sS1[TILE];

    const int CL = LL / NC;
    int gid  = blockIdx.x;
    const int dblk = gid % (DD / 256); gid /= (DD / 256);
    const int c    = gid % NC;
    const int b    = gid / NC;
    const int d    = dblk * 256 + threadIdx.x;

    float an[NN], h[NN];
    const size_t ebase = (((size_t)b * NC + c) * DD + d) * NN;
    #pragma unroll
    for (int n = 0; n < NN; ++n) {
        an[n] = -expf(A_log[(size_t)d * NN + n]);
        h[n]  = E ? E[ebase + n] : 0.0f;
    }
    const float wd  = Wd[d];
    const float bdv = bd[d];

    const size_t l0g = (size_t)b * LL + (size_t)c * CL;
    const float* xp  = x + l0g * DD + d;
    float*       yp  = y + l0g * DD + d;

    for (int t0 = 0; t0 < CL; t0 += TILE) {
        const float* bsrc = Bp + (l0g + t0) * NN;
        const float* csrc = Cc + (l0g + t0) * NN;
        for (int s = threadIdx.x; s < TILE * NN / 4; s += 256) {
            g2l_b128(bsrc + 4 * s, sBp + 4 * s);
            g2l_b128(csrc + 4 * s, sC  + 4 * s);
        }
        if (threadIdx.x < TILE / 4)
            g2l_b128(s1 + l0g + t0 + 4 * threadIdx.x, sS1 + 4 * threadIdx.x);
        async_wait0();
        __syncthreads();

        for (int l = 0; l < TILE; ++l) {
            const int lg = t0 + l;
            if (lg + 8 < CL) __builtin_prefetch(xp + (size_t)(lg + 8) * DD, 0, 0);
            const float xv    = xp[(size_t)lg * DD];
            const float delta = softplusf(fmaf(sS1[l], wd, bdv));
            const float dx    = delta * xv;
            float yv = 0.0f;
            #pragma unroll
            for (int n = 0; n < NN; ++n) {
                const float dA = expf(delta * an[n]);
                const float u  = dx * sBp[l * NN + n];
                h[n] = fmaf(dA, h[n], u);
                yv   = fmaf(h[n], sC[l * NN + n], yv);
            }
            yp[(size_t)lg * DD] = yv;
        }
        __syncthreads();   // before next tile's LDS overwrite
    }
}

// ---------------------------------------------------------------------------
extern "C" void kernel_launch(void* const* d_in, const int* in_sizes, int n_in,
                              void* d_out, int out_size, void* d_ws, size_t ws_size,
                              hipStream_t stream)
{
    const float* x     = (const float*)d_in[0];
    const float* A_log = (const float*)d_in[1];
    const float* Wb    = (const float*)d_in[2];
    const float* bb    = (const float*)d_in[3];
    const float* Wc    = (const float*)d_in[4];
    const float* bc    = (const float*)d_in[5];
    const float* W1    = (const float*)d_in[6];
    const float* b1    = (const float*)d_in[7];
    const float* Wd    = (const float*)d_in[8];
    const float* bd    = (const float*)d_in[9];
    float* y = (float*)d_out;

    // ---- workspace layout (deterministic: depends only on ws_size) --------
    const size_t f = sizeof(float);
    char* w = (char*)d_ws;
    float* wsBp = (float*)w; w += (size_t)BB * LL * NN * f;
    float* wsC  = (float*)w; w += (size_t)BB * LL * NN * f;
    float* wsS1 = (float*)w; w += (size_t)BB * LL * f;
    const size_t used = (size_t)(w - (char*)d_ws);

    const size_t per_chunk = 3ull * BB * DD * NN * f;   // P + S + E per chunk
    int NC = 1;
    for (int cand = 16; cand >= 2; cand >>= 1) {
        if (used + per_chunk * (size_t)cand <= ws_size) { NC = cand; break; }
    }
    float *P = nullptr, *S = nullptr, *E = nullptr;
    if (NC > 1) {
        P = (float*)w; w += (size_t)BB * NC * DD * NN * f;
        S = (float*)w; w += (size_t)BB * NC * DD * NN * f;
        E = (float*)w; w += (size_t)BB * NC * DD * NN * f;
    }

    // ---- 1) projections via WMMA (64 blocks x 8 waves x 16-row tiles) -----
    proj_kernel<<<(BB * LL) / 128, 256, 0, stream>>>(
        x, Wb, bb, Wc, bc, W1, b1, wsBp, wsC, wsS1);

    // ---- 2) chunked diagonal scan -----------------------------------------
    const int scan_blocks = BB * NC * (DD / 256);
    if (NC > 1) {
        scan_phase1<<<scan_blocks, 256, 0, stream>>>(
            x, A_log, Wd, bd, wsBp, wsS1, P, S, NC);
        scan_phase2<<<(BB * DD * NN + 255) / 256, 256, 0, stream>>>(P, S, E, NC);
    }
    scan_phase3<<<scan_blocks, 256, 0, stream>>>(
        x, A_log, Wd, bd, wsBp, wsC, wsS1, E, y, NC);
}
